// GridModule_15736760172714
// MI455X (gfx1250) — matrix operand order repaired and checked by reading, not compiled, not run on hardware
//
#include <hip/hip_runtime.h>

// ---------------------------------------------------------------------------
// Sizes from the reference
// ---------------------------------------------------------------------------
#define T_STEPS 1024
#define I_DIM   128
#define H_DIM   512
#define O_DIM   512
#define M_PAT   32768
#define PAT_W   1024          // O + H
#define BETA    8.0f
#define KSPLIT  4             // key-dimension split across workgroups
#define KEYS_PER_PART (M_PAT / KSPLIT)   // 8192
#define KB_KEYS 64            // keys per LDS block
#define NBLK    (KEYS_PER_PART / KB_KEYS) // 128
#define QT      16            // query rows per workgroup (one WMMA M tile)

typedef unsigned short u16;
typedef unsigned int   u32;
typedef unsigned long long u64;
typedef __attribute__((ext_vector_type(8)))  u16    u16x8;
typedef __attribute__((ext_vector_type(8)))  float  v8f;
typedef __attribute__((ext_vector_type(16))) __bf16 v16bf;
typedef __attribute__((ext_vector_type(4)))  u32    v4u;
typedef __attribute__((ext_vector_type(8)))  int    v8i;
typedef __attribute__((ext_vector_type(4)))  int    v4i;

union V16U { u16x8 h[2]; u16 s[16]; v16bf v; };

__device__ __forceinline__ u16 f2bf(float f) {
  u32 u = __builtin_bit_cast(u32, f);
  u32 r = u + 0x7FFFu + ((u >> 16) & 1u);   // round-to-nearest-even
  return (u16)(r >> 16);
}

__device__ __forceinline__ v8f wmma_bf16(v16bf a, v16bf b, v8f c) {
  // (neg_a, A, neg_b, B, c_mod, C, reuse_a, reuse_b)
  return __builtin_amdgcn_wmma_f32_16x16x32_bf16(false, a, false, b, (short)0, c,
                                                 false, false);
}

// A operand (16x32 bf16), row-major source in LDS with given element stride.
__device__ __forceinline__ v16bf load_a(const u16* base, int stride, int row,
                                        int hf, int k0) {
  V16U t;
  const u16* p = base + row * stride + k0 + hf * 8;
  t.h[0] = *(const u16x8*)p;          // K = k0 + hf*8 .. +7
  t.h[1] = *(const u16x8*)(p + 16);   // K = k0 + 16 + hf*8 .. +7
  return t.v;
}

// B operand (32x16 bf16): lane n (=lane&15) supplies column n, K = k0+hf*8..+7, k0+16+hf*8..+7.
__device__ __forceinline__ v16bf load_b(const u16* base, int stride, int n,
                                        int hf, int k0) {
  V16U t;
  const u16* p = base + n * stride + k0 + hf * 8;
  t.h[0] = *(const u16x8*)p;
  t.h[1] = *(const u16x8*)(p + 16);
  return t.v;
}

// CDNA5 async global->LDS copy (16B per lane), tracked by ASYNCcnt.
__device__ __forceinline__ void async_b128(u32 lds_off, u32 goff, const void* sbase) {
  asm volatile("global_load_async_to_lds_b128 %0, %1, %2"
               :: "v"(lds_off), "v"(goff), "s"(sbase) : "memory");
}
__device__ __forceinline__ void wait_async() {
  asm volatile("s_wait_asynccnt 0" ::: "memory");
}

// CDNA5 Tensor Data Mover: 2D tile load, descriptor groups per ISA 8.3/8.4.
// g1 carries {data_size, tensor_dim0/1, tile_dim0/1, dim0_stride}; groups 2/3
// (and the trailing group) are zero (2D tensor, tile_dim2=0 -> unused).
__device__ __forceinline__ void tdm_load_2d(u32 lds_addr, const void* gptr,
                                            v8i g1) {
  u64 ga = (u64)gptr;
  v4u g0;
  g0[0] = 1u;                              // count=1, no gather, user mode
  g0[1] = lds_addr;                        // LDS byte address
  g0[2] = (u32)ga;                         // global_addr[31:0]
  g0[3] = (u32)(ga >> 32) | (2u << 30);    // global_addr[56:32] | type=2
  v4i z4 = {0, 0, 0, 0};
  v8i z8 = {0, 0, 0, 0, 0, 0, 0, 0};
  __builtin_amdgcn_tensor_load_to_lds(g0, g1, z4, z4, z8, 0);
}
__device__ __forceinline__ void wait_tensor() {
  __builtin_amdgcn_s_wait_tensorcnt((short)0);
}

// ---------------------------------------------------------------------------
// Kernel 1: patterns -> bf16 keys (row-major), values (row-major + col-major),
// plus key/value squared norms.
// ---------------------------------------------------------------------------
__global__ __launch_bounds__(128) void prep_patterns(
    const float* __restrict__ patterns, u16* __restrict__ Kb,
    u16* __restrict__ Vb, u16* __restrict__ VbT,
    float* __restrict__ norK, float* __restrict__ norV) {
  __shared__ float sK, sV;
  const int m = blockIdx.x;
  const int tid = threadIdx.x;
  if (tid == 0) { sK = 0.f; sV = 0.f; }
  __syncthreads();
  const float* prow = patterns + (size_t)m * PAT_W;
  float nv = 0.f, nk = 0.f;
  for (int c = tid; c < O_DIM; c += 128) {
    float xv = prow[c];                       // x_vals = patterns[:, :O]
    u16 b = f2bf(xv);
    Vb[m * O_DIM + c] = b;
    VbT[(size_t)c * M_PAT + m] = b;           // [col][key] for B operand
    nv += xv * xv;
    float gk = prow[O_DIM + c];               // g_keys = patterns[:, O:]
    Kb[m * H_DIM + c] = f2bf(gk);
    nk += gk * gk;
  }
  atomicAdd(&sV, nv);
  atomicAdd(&sK, nk);
  __syncthreads();
  if (tid == 0) { norV[m] = sV; norK[m] = sK; }
}

// ---------------------------------------------------------------------------
// Kernel 2: f32 -> bf16 grid-stride convert (x_obs queries)
// ---------------------------------------------------------------------------
__global__ void bf16_convert(const float* __restrict__ in, u16* __restrict__ out,
                             int n) {
  for (int i = blockIdx.x * blockDim.x + threadIdx.x; i < n;
       i += gridDim.x * blockDim.x)
    out[i] = f2bf(in[i]);
}

// ---------------------------------------------------------------------------
// Kernel 3: pre[t][j] = inp[t] . W_ih[j] + b_ih[j] + b_hh[j]
// ---------------------------------------------------------------------------
__global__ __launch_bounds__(128) void inp_gemm(
    const float* __restrict__ inp, const float* __restrict__ W_ih,
    const float* __restrict__ b_ih, const float* __restrict__ b_hh,
    float* __restrict__ pre) {
  __shared__ float xr[I_DIM];
  const int t = blockIdx.x, tid = threadIdx.x;
  if (tid < I_DIM) xr[tid] = inp[t * I_DIM + tid];
  __syncthreads();
  for (int j = tid; j < H_DIM; j += 128) {
    const float* wr = W_ih + j * I_DIM;
    float s = 0.f;
    #pragma unroll 8
    for (int k = 0; k < I_DIM; k += 4) {
      float4 w4 = *(const float4*)(wr + k);
      s += w4.x * xr[k] + w4.y * xr[k + 1] + w4.z * xr[k + 2] + w4.w * xr[k + 3];
    }
    pre[t * H_DIM + j] = s + b_ih[j] + b_hh[j];
  }
}

// ---------------------------------------------------------------------------
// Kernel 4: sequential recurrence (latency-bound). One workgroup, h in LDS.
// ---------------------------------------------------------------------------
__global__ __launch_bounds__(512) void rnn_scan(
    const float* __restrict__ pre, const float* __restrict__ W_hh,
    const float* __restrict__ h0, float* __restrict__ g_out,
    u16* __restrict__ Qg, float* __restrict__ h_last) {
  __shared__ float hs[H_DIM];
  const int j = threadIdx.x;
  hs[j] = h0[j];
  __syncthreads();
  const float* wr = W_hh + j * H_DIM;
  for (int t = 0; t < T_STEPS; ++t) {
    float s = pre[t * H_DIM + j];
    #pragma unroll 8
    for (int k = 0; k < H_DIM; k += 4) {
      float4 w4 = *(const float4*)(wr + k);
      s += w4.x * hs[k] + w4.y * hs[k + 1] + w4.z * hs[k + 2] + w4.w * hs[k + 3];
    }
    float h = tanhf(s);
    g_out[t * H_DIM + j] = h;
    Qg[t * H_DIM + j] = f2bf(h);
    __syncthreads();
    hs[j] = h;
    __syncthreads();
  }
  h_last[j] = hs[j];
}

// ---------------------------------------------------------------------------
// Kernel 5: fused retrieval, TDM double-buffered.
//   exponent = beta*(2 q.k - |k|^2 - |q|^2) <= 0  => no online max needed.
// Block = 128 threads (4 waves), one 16-query tile, KSPLIT key partitions.
// LDS: Qs 16K | Ks[2] 2x64K | Vt[2] 2x64K | Ps 2K | qn  (~274 KB of 320 KB)
// ---------------------------------------------------------------------------
#define LOFF_KS  16384u
#define LOFF_VT  (16384u + 131072u)
#define LOFF_PS  (16384u + 262144u)
#define LOFF_QN  (16384u + 262144u + 2048u)
#define SMEM_SZ  (16384 + 262144 + 2048 + 64)

__global__ __launch_bounds__(128) void retrieve_kernel(
    const u16* __restrict__ Qb, const float* __restrict__ Qf,
    const u16* __restrict__ KeyRM, const u16* __restrict__ VT,
    const float* __restrict__ norKey, float* __restrict__ accp,
    float* __restrict__ sump) {
  extern __shared__ char smem[];
  u16* Qs  = (u16*)smem;
  u16* Ps  = (u16*)(smem + LOFF_PS);
  float* qn = (float*)(smem + LOFF_QN);

  const int tid  = threadIdx.x;
  const int lane = tid & 31;
  const int w    = tid >> 5;     // wave id 0..3 (uniform per wave)
  const int row  = lane & 15;    // A-row / B-column index
  const int hf   = lane >> 4;    // K-half selector

  const int qt       = blockIdx.x & 63;
  const int part     = blockIdx.x >> 6;
  const int qbase    = qt * QT;
  const int keybase0 = part * KEYS_PER_PART;

  const u32 lds0 = (u32)(size_t)smem;

  // TDM group-1 descriptors (uniform).
  // Keys tensor: [M_PAT][512] row-major; tile 512(x) x 64(y) -> LDS [key][feat]
  const v8i g1k = {0x00010000, 0x02000000, (int)0x80000000u, 0x02000000,
                   64, 512, 0, 0};
  // Values tensor (VbT): [512][M_PAT] row-major; tile 64(x) x 512(y)
  //   -> LDS [col][key]  (exactly the B-operand layout)
  const v8i g1v = {0x00010000, (int)0x80000000u, 0x02000000, 0x00400000,
                   512, 32768, 0, 0};

  // Stage Q tile (16KB bf16) with async-to-LDS; uniform, fully unrolled.
  #pragma unroll
  for (int it = 0; it < 8; ++it) {
    int j = tid + it * 128;
    async_b128(lds0 + (u32)(j * 16), (u32)((qbase * 512 + j * 8) * 2), Qb);
  }
  if (tid < 16) qn[tid] = 0.f;
  __syncthreads();
  {  // query norms from f32 source (the stable softmax shift)
    int r = tid >> 3, p = tid & 7;
    const float* q = Qf + (qbase + r) * 512 + p * 64;
    float s = 0.f;
    #pragma unroll 16
    for (int k = 0; k < 64; ++k) s += q[k] * q[k];
    atomicAdd(&qn[r], s);
  }
  wait_async();
  __syncthreads();

  float qnv[8];
  #pragma unroll
  for (int g = 0; g < 8; ++g) qnv[g] = qn[g + 8 * hf];

  V16U onesu;
  #pragma unroll
  for (int i = 0; i < 16; ++i) onesu.s[i] = 0x3F80;  // bf16 1.0
  const v16bf ones = onesu.v;

  v8f acc[8], Csum;
  #pragma unroll
  for (int ct = 0; ct < 8; ++ct)
    #pragma unroll
    for (int g = 0; g < 8; ++g) acc[ct][g] = 0.f;
  #pragma unroll
  for (int g = 0; g < 8; ++g) Csum[g] = 0.f;

  // Prologue: wave 0 DMAs first tiles into buffer 0.
  if (w == 0) {
    tdm_load_2d(lds0 + LOFF_KS, KeyRM + (size_t)keybase0 * 512, g1k);
    tdm_load_2d(lds0 + LOFF_VT, VT + keybase0, g1v);
  }

  for (int kb = 0; kb < NBLK; ++kb) {
    const int p = kb & 1;
    const int keybase = keybase0 + kb * KB_KEYS;
    const u16* Ksp = (const u16*)(smem + LOFF_KS + (u32)p * 65536u);
    const u16* Vtp = (const u16*)(smem + LOFF_VT + (u32)p * 65536u);

    if (w == 0) wait_tensor();     // current buffer's DMA complete
    __syncthreads();               // everyone done with the other buffer too
    if (w == 0 && kb + 1 < NBLK) { // prefetch next tiles into other buffer
      const int nb = keybase0 + (kb + 1) * KB_KEYS;
      tdm_load_2d(lds0 + LOFF_KS + (u32)(p ^ 1) * 65536u,
                  KeyRM + (size_t)nb * 512, g1k);
      tdm_load_2d(lds0 + LOFF_VT + (u32)(p ^ 1) * 65536u, VT + nb, g1v);
    }

    // Scores: wave w owns keys [w*16, w*16+16).  S = Q(16x512) . K^T
    v8f S;
    #pragma unroll
    for (int g = 0; g < 8; ++g) S[g] = 0.f;
    const u16* kb_l = Ksp + (w * 16) * 512;
    {
      v16bf a = load_a(Qs, 512, row, hf, 0);
      v16bf b = load_b(kb_l, 512, row, hf, 0);
      #pragma unroll
      for (int k0 = 0; k0 < 512; k0 += 32) {
        v16bf an = a, bn = b;
        if (k0 + 32 < 512) {                     // prefetch next operands
          an = load_a(Qs, 512, row, hf, k0 + 32);
          bn = load_b(kb_l, 512, row, hf, k0 + 32);
        }
        S = wmma_bf16(a, b, S);
        a = an; b = bn;
      }
    }

    const float nk = norKey[keybase + w * 16 + row];
    #pragma unroll
    for (int g = 0; g < 8; ++g) {
      float e = __expf(BETA * (2.0f * S[g] - nk - qnv[g]));  // <= 1 always
      Ps[(g + 8 * hf) * 64 + w * 16 + row] = f2bf(e);
    }
    __syncthreads();  // full P tile visible

    // acc(16 x 128 cols per wave) += P(16x64) @ V(64x512); wave0 rowsums.
    #pragma unroll
    for (int kc = 0; kc < 64; kc += 32) {
      v16bf a = load_a(Ps, 64, row, hf, kc);
      if (w == 0) Csum = wmma_bf16(a, ones, Csum);
      v16bf bc = load_b(Vtp + (w * 128) * 64, 64, row, hf, kc);
      #pragma unroll
      for (int ct = 0; ct < 8; ++ct) {
        v16bf bn = bc;
        if (ct < 7)                               // prefetch next B tile
          bn = load_b(Vtp + (w * 128 + (ct + 1) * 16) * 64, 64, row, hf, kc);
        acc[ct] = wmma_bf16(a, bc, acc[ct]);
        bc = bn;
      }
    }
  }

  // Write partial accumulators / partial row sums (deterministic reduction)
  const int cbase = w * 128;
  #pragma unroll
  for (int ct = 0; ct < 8; ++ct)
    #pragma unroll
    for (int g = 0; g < 8; ++g) {
      int m = g + 8 * hf;
      accp[part * (1024 * 512) + (qbase + m) * 512 + cbase + ct * 16 + row] =
          acc[ct][g];
    }
  if (w == 0 && row == 0) {  // lanes 0 and 16 carry column 0 of Csum
    #pragma unroll
    for (int g = 0; g < 8; ++g)
      sump[part * 1024 + qbase + g + 8 * hf] = Csum[g];
  }
}

// ---------------------------------------------------------------------------
// Kernel 6: reduce KSPLIT partials and normalize by softmax denominator
// ---------------------------------------------------------------------------
__global__ void normalize_kernel(const float* __restrict__ accp,
                                 const float* __restrict__ sump,
                                 float* __restrict__ out) {
  int idx = blockIdx.x * blockDim.x + threadIdx.x;
  if (idx >= 1024 * 512) return;
  int rowq = idx >> 9;
  float a = 0.f, s = 0.f;
  #pragma unroll
  for (int p = 0; p < KSPLIT; ++p) {
    a += accp[p * (1024 * 512) + idx];
    s += sump[p * 1024 + rowq];
  }
  out[idx] = a / s;
}

// ---------------------------------------------------------------------------
extern "C" void kernel_launch(void* const* d_in, const int* in_sizes, int n_in,
                              void* d_out, int out_size, void* d_ws, size_t ws_size,
                              hipStream_t stream) {
  const float* inp         = (const float*)d_in[0];
  const float* x_obs       = (const float*)d_in[1];
  const float* patterns    = (const float*)d_in[2];
  const float* last_hidden = (const float*)d_in[3];
  const float* W_ih        = (const float*)d_in[4];
  const float* W_hh        = (const float*)d_in[5];
  const float* b_ih        = (const float*)d_in[6];
  const float* b_hh        = (const float*)d_in[7];
  float* out = (float*)d_out;

  // d_out layout (floats): prob_g | prob_x | g_cur | h_last
  const size_t OUT_PG = 0, OUT_PX = 524288, OUT_G = 1048576, OUT_HL = 1572864;

  // workspace carve (bytes)
  char* ws = (char*)d_ws;
  float* pre  = (float*)(ws + 0);                 //  2 MB
  u16*   Qg   = (u16*)(ws + (2u << 20));          //  1 MB
  u16*   Qx   = (u16*)(ws + (3u << 20));          //  1 MB
  float* norK = (float*)(ws + (4u << 20));        // 128 KB
  float* norV = (float*)(ws + (4u << 20) + 131072u);
  u16*   Kb   = (u16*)(ws + (4u << 20) + 262144u);             // 32 MB
  u16*   Vb   = (u16*)(ws + (4u << 20) + 262144u + (32u << 20));
  u16*   VbT  = (u16*)(ws + (4u << 20) + 262144u + (64u << 20));
  float* accG = (float*)(ws + (4u << 20) + 262144u + (96u << 20));          // 8 MB
  float* accX = (float*)(ws + (4u << 20) + 262144u + (96u << 20) + (8u << 20));
  float* sumG = (float*)(ws + (4u << 20) + 262144u + (96u << 20) + (16u << 20));
  float* sumX = (float*)(ws + (4u << 20) + 262144u + (96u << 20) + (16u << 20) + 16384u);

  prep_patterns<<<M_PAT, 128, 0, stream>>>(patterns, Kb, Vb, VbT, norK, norV);
  bf16_convert<<<2048, 256, 0, stream>>>(x_obs, Qx, T_STEPS * O_DIM);
  inp_gemm<<<T_STEPS, 128, 0, stream>>>(inp, W_ih, b_ih, b_hh, pre);
  rnn_scan<<<1, 512, 0, stream>>>(pre, W_hh, last_hidden, out + OUT_G, Qg,
                                  out + OUT_HL);

  retrieve_kernel<<<64 * KSPLIT, 128, SMEM_SZ, stream>>>(Qg, out + OUT_G, Kb,
                                                         VbT, norK, accG, sumG);
  retrieve_kernel<<<64 * KSPLIT, 128, SMEM_SZ, stream>>>(Qx, x_obs, Vb, VbT,
                                                         norV, accX, sumX);
  normalize_kernel<<<2048, 256, 0, stream>>>(accG, sumG, out + OUT_PG);
  normalize_kernel<<<2048, 256, 0, stream>>>(accX, sumX, out + OUT_PX);
  (void)in_sizes; (void)n_in; (void)out_size; (void)ws_size;
}
